// PopulationGAT_32641751450120
// MI455X (gfx1250) — compile-verified, matrix operand order checked by reading, not run on hardware
//
#include <hip/hip_runtime.h>

typedef _Float16 half_t;
typedef __attribute__((ext_vector_type(16))) _Float16 v16h;
typedef __attribute__((ext_vector_type(8)))  _Float16 v8h;
typedef __attribute__((ext_vector_type(8)))  float    v8f;

#define EMBED   64
#define LSTM_H  128
#define T_SEQ   20
#define HEADS   2
#define GAT_C   32
#define OUTC    64      // HEADS*GAT_C
#define OUT_DIM 10

// ---------------- branchless activations (keep EXEC all-1s around WMMA) ----------------

__device__ __forceinline__ float fast_rcp(float x)   { return __builtin_amdgcn_rcpf(x); }
__device__ __forceinline__ float fast_sigmoid(float x) {
    return fast_rcp(1.f + __expf(-x));                 // v_exp_f32 + v_rcp_f32, no branches
}
__device__ __forceinline__ float fast_tanh(float x) {
#if __has_builtin(__builtin_amdgcn_tanhf)
    return __builtin_amdgcn_tanhf(x);                  // V_TANH_F32 (gfx1250 TRANS op)
#else
    return 2.f * fast_rcp(1.f + __expf(-2.f * x)) - 1.f;
#endif
}

// ---------------- WMMA helpers ----------------

__device__ __forceinline__ v8f wmma_f16(v16h a, v16h b, v8f c) {
    return __builtin_amdgcn_wmma_f32_16x16x32_f16(
        /*neg_a=*/false, a, /*neg_b=*/false, b,
        /*c_mod=*/(short)0, c, /*reuse_a=*/false, /*reuse_b=*/false);
}

__device__ __forceinline__ v16h make_frag(const half_t* p0, const half_t* p1) {
    v8h a = *(const v8h*)p0;
    v8h b = *(const v8h*)p1;
    v16h r;
#pragma unroll
    for (int i = 0; i < 8; ++i) { r[i] = a[i]; r[8 + i] = b[i]; }
    return r;
}

// A-matrix fragment (16x32 f16, ISA 7.12.2 layout) from a row-major [16][stride] buffer.
__device__ __forceinline__ v16h load_A_frag(const half_t* base, int stride, int kc) {
    int lane = threadIdx.x & 31;
    int m  = lane & 15;
    int kb = ((lane < 16) ? 0 : 8) + kc * 32;
    const half_t* p = base + m * stride + kb;
    return make_frag(p, p + 16);   // K = kb..kb+7 , kb+16..kb+23
}

// B-matrix fragment from pre-packed fragment-ordered array: 32B contiguous per lane.
__device__ __forceinline__ v16h load_B_frag(const half_t* base, int fragidx) {
    const half_t* p = base + ((size_t)fragidx * 32 + (threadIdx.x & 31)) * 16;
    return make_frag(p, p + 8);
}

// ---------------- precompute kernels ----------------

__global__ void convert_f16(const float* src, half_t* dst, int n) {
    int i = blockIdx.x * 256 + threadIdx.x;
    if (i < n) dst[i] = (half_t)src[i];
}

// Pack weights into B-fragment order.
// transpose=1: W is [outCols][K] (LSTM weights, row stride ldW=K)  -> B(k,n)=W[n][k]
// transpose=0: W is [K][outCols] (GAT weights, row stride ldW=cols)-> B(k,n)=W[k][n]
__global__ void pack_b_frags(const float* W, half_t* out, int K, int nct, int ldW, int transpose) {
    int idx = blockIdx.x * 256 + threadIdx.x;
    int nkc = K >> 5;
    int total = nct * nkc * 512;
    if (idx >= total) return;
    int e    = idx & 15;
    int lane = (idx >> 4) & 31;
    int kc   = (idx >> 9) % nkc;
    int ct   = idx / (nkc * 512);
    int n  = lane & 15;
    int kb = (lane < 16) ? 0 : 8;
    int k  = kc * 32 + kb + (e < 8 ? e : 8 + e);
    float v = transpose ? W[(ct * 16 + n) * ldW + k] : W[k * ldW + (ct * 16 + n)];
    out[idx] = (half_t)v;
}

// ---------------- LSTM kernel (16 nodes / block, 8 waves) ----------------

__global__ __launch_bounds__(256) void lstm_kernel(
    const int* __restrict__ x, const int* __restrict__ lengths,
    const half_t* __restrict__ emb16,
    const half_t* __restrict__ fih_f, const half_t* __restrict__ fhh_f,
    const float*  __restrict__ b_f,
    const half_t* __restrict__ fih_b, const float* __restrict__ b_b,
    half_t* __restrict__ feat /* [N][256] f16 */)
{
    __shared__ half_t xh[16 * EMBED];
    __shared__ half_t hh[16 * LSTM_H];
    __shared__ int    lenl[16];

    const int tid  = threadIdx.x;
    const int wave = tid >> 5, lane = tid & 31;
    const int nodeBase = blockIdx.x * 16;
    const int n     = lane & 15;
    const int mBase = (lane < 16) ? 0 : 8;
    const int hcol  = wave * 16 + n;

    if (tid < 16) lenl[tid] = lengths[nodeBase + tid];
    for (int i = tid; i < 16 * LSTM_H; i += 256) hh[i] = (half_t)0.f;

    float c_reg[8], h_reg[8];
#pragma unroll
    for (int e = 0; e < 8; ++e) { c_reg[e] = 0.f; h_reg[e] = 0.f; }

    float bias[4];
#pragma unroll
    for (int g = 0; g < 4; ++g) bias[g] = b_f[g * 128 + hcol];

    const int nd  = tid >> 4;     // staging role: node 0..15
    const int seg = tid & 15;     // staging role: 4-half segment

    for (int t = 0; t < T_SEQ; ++t) {
        // stage x_t tile into LDS (f16)
        {
            int tok = x[(nodeBase + nd) * T_SEQ + t];
            *(unsigned long long*)(&xh[nd * EMBED + seg * 4]) =
                *(const unsigned long long*)(emb16 + (size_t)tok * EMBED + seg * 4);
        }
        __syncthreads();

        v16h ax0 = load_A_frag(xh, EMBED, 0);
        v16h ax1 = load_A_frag(xh, EMBED, 1);
        v16h ah0 = load_A_frag(hh, LSTM_H, 0);
        v16h ah1 = load_A_frag(hh, LSTM_H, 1);
        v16h ah2 = load_A_frag(hh, LSTM_H, 2);
        v16h ah3 = load_A_frag(hh, LSTM_H, 3);

        v8f acc[4];
#pragma unroll
        for (int g = 0; g < 4; ++g) {
            int ct = g * 8 + wave;
            v8f a = {};
            a = wmma_f16(ax0, load_B_frag(fih_f, ct * 2 + 0), a);
            a = wmma_f16(ax1, load_B_frag(fih_f, ct * 2 + 1), a);
            a = wmma_f16(ah0, load_B_frag(fhh_f, ct * 4 + 0), a);
            a = wmma_f16(ah1, load_B_frag(fhh_f, ct * 4 + 1), a);
            a = wmma_f16(ah2, load_B_frag(fhh_f, ct * 4 + 2), a);
            a = wmma_f16(ah3, load_B_frag(fhh_f, ct * 4 + 3), a);
            acc[g] = a;
        }

#pragma unroll
        for (int e = 0; e < 8; ++e) {
            float gi = acc[0][e] + bias[0];
            float gf = acc[1][e] + bias[1];
            float gg = acc[2][e] + bias[2];
            float go = acc[3][e] + bias[3];
            float i_ = fast_sigmoid(gi);
            float f_ = fast_sigmoid(gf);
            float o_ = fast_sigmoid(go);
            float cn = f_ * c_reg[e] + i_ * fast_tanh(gg);
            float hn = o_ * fast_tanh(cn);
            bool act = (t < lenl[mBase + e]);
            c_reg[e] = act ? cn : c_reg[e];
            h_reg[e] = act ? hn : h_reg[e];
        }
        __syncthreads();   // all reads of xh/hh done

        // publish h for next step
#pragma unroll
        for (int e = 0; e < 8; ++e)
            hh[(mBase + e) * LSTM_H + hcol] = (half_t)h_reg[e];
    }

    // write h_fwd into feat[:, 0:128]
#pragma unroll
    for (int e = 0; e < 8; ++e)
        feat[(size_t)(nodeBase + mBase + e) * 256 + hcol] = (half_t)h_reg[e];

    // ---- backward single cell: gates = x_last @ Wih_b^T + b_b  (h=0,c=0) ----
    {
        int len = lenl[nd];
        int tok = x[(nodeBase + nd) * T_SEQ + (len - 1)];
        *(unsigned long long*)(&xh[nd * EMBED + seg * 4]) =
            *(const unsigned long long*)(emb16 + (size_t)tok * EMBED + seg * 4);
    }
    __syncthreads();

    v16h bx0 = load_A_frag(xh, EMBED, 0);
    v16h bx1 = load_A_frag(xh, EMBED, 1);
    v8f acc[4];
#pragma unroll
    for (int g = 0; g < 4; ++g) {
        int ct = g * 8 + wave;
        v8f a = {};
        a = wmma_f16(bx0, load_B_frag(fih_b, ct * 2 + 0), a);
        a = wmma_f16(bx1, load_B_frag(fih_b, ct * 2 + 1), a);
        acc[g] = a;
    }
#pragma unroll
    for (int e = 0; e < 8; ++e) {
        float gi = acc[0][e] + b_b[0 * 128 + hcol];
        float gg = acc[2][e] + b_b[2 * 128 + hcol];
        float go = acc[3][e] + b_b[3 * 128 + hcol];
        float i_ = fast_sigmoid(gi);
        float o_ = fast_sigmoid(go);
        float cn = i_ * fast_tanh(gg);      // f*c0 = 0
        float hb = o_ * fast_tanh(cn);
        feat[(size_t)(nodeBase + mBase + e) * 256 + 128 + hcol] = (half_t)hb;
    }
}

// ---------------- GAT linear (WMMA): H = A(f16,[N][K]) @ W -> f32 [N][64] ----------------

__global__ __launch_bounds__(128) void gat_linear(
    const half_t* __restrict__ A, const half_t* __restrict__ Bfrag,
    float* __restrict__ H, int K)
{
    const int wave = threadIdx.x >> 5, lane = threadIdx.x & 31;
    const int nodeBase = blockIdx.x * 16;
    const int n = lane & 15;
    const int mBase = (lane < 16) ? 0 : 8;
    const int nkc = K >> 5;

    v8f acc = {};
    for (int kc = 0; kc < nkc; ++kc) {
        const half_t* p = A + (size_t)(nodeBase + n) * K + kc * 32 + ((lane < 16) ? 0 : 8);
        v16h a = make_frag(p, p + 16);
        acc = wmma_f16(a, load_B_frag(Bfrag, wave * nkc + kc), acc);
    }
#pragma unroll
    for (int e = 0; e < 8; ++e)
        H[(size_t)(nodeBase + mBase + e) * OUTC + wave * 16 + n] = acc[e];
}

// ---------------- attention helpers ----------------

__device__ __forceinline__ unsigned fenc(float f) {
    unsigned u = __float_as_uint(f);
    return (u & 0x80000000u) ? ~u : (u | 0x80000000u);
}
__device__ __forceinline__ float fdec(unsigned u) {
    return (u & 0x80000000u) ? __uint_as_float(u ^ 0x80000000u) : __uint_as_float(~u);
}

__global__ void node_scores(const float* __restrict__ H,
                            const float* __restrict__ a_src, const float* __restrict__ a_dst,
                            float* __restrict__ ssrc, float* __restrict__ sdst, int N)
{
    int idx = blockIdx.x * 256 + threadIdx.x;      // (node, head)
    if (idx >= N * HEADS) return;
    int nd = idx >> 1, h = idx & 1;
    float as = 0.f, ad = 0.f;
    const float* row = H + (size_t)nd * OUTC + h * GAT_C;
#pragma unroll
    for (int c = 0; c < GAT_C; ++c) {
        as += row[c] * a_src[h * GAT_C + c];
        ad += row[c] * a_dst[h * GAT_C + c];
    }
    ssrc[idx] = as; sdst[idx] = ad;
}

__global__ void fill_u32(unsigned* p, long n) {
    long i = (long)blockIdx.x * 256 + threadIdx.x;
    if (i < n) p[i] = 0u;
}

__global__ void edge_max(const int* __restrict__ ei, int nE, int N,
                         const float* __restrict__ ssrc, const float* __restrict__ sdst,
                         unsigned* __restrict__ mbuf)
{
    int e = blockIdx.x * 256 + threadIdx.x;
    if (e >= nE + N) return;
    int s, d;
    if (e < nE) { s = ei[e]; d = ei[nE + e]; } else { s = d = e - nE; }
#pragma unroll
    for (int h = 0; h < HEADS; ++h) {
        float v = ssrc[s * 2 + h] + sdst[d * 2 + h];
        v = (v > 0.f) ? v : 0.2f * v;                       // LeakyReLU(0.2)
        atomicMax(&mbuf[d * 2 + h], fenc(v));
    }
}

__global__ void edge_acc(const int* __restrict__ ei, int nE, int N,
                         const float* __restrict__ ssrc, const float* __restrict__ sdst,
                         const unsigned* __restrict__ mbuf, const float* __restrict__ H,
                         float* __restrict__ agg, float* __restrict__ den)
{
    int idx = blockIdx.x * 256 + threadIdx.x;      // (edge, head)
    int e = idx >> 1, h = idx & 1;
    if (e >= nE + N) return;
    int s, d;
    if (e < nE) { s = ei[e]; d = ei[nE + e]; } else { s = d = e - nE; }
    float v = ssrc[s * 2 + h] + sdst[d * 2 + h];
    v = (v > 0.f) ? v : 0.2f * v;
    float ex = __expf(v - fdec(mbuf[d * 2 + h]));
    atomicAdd(&den[d * 2 + h], ex);
    const float* hs = H + (size_t)s * OUTC + h * GAT_C;
    float* ag = agg + (size_t)d * OUTC + h * GAT_C;
#pragma unroll 4
    for (int c = 0; c < GAT_C; ++c) atomicAdd(&ag[c], ex * hs[c]);
}

// normalize + bias + relu, emit f16 (A-matrix for next layer)
__global__ void norm_relu16(const float* __restrict__ agg, const float* __restrict__ den,
                            const float* __restrict__ bias, half_t* __restrict__ out16, int N)
{
    int idx = blockIdx.x * 256 + threadIdx.x;
    if (idx >= N * OUTC) return;
    int nd = idx >> 6, c = idx & 63;
    float v = agg[idx] / den[nd * 2 + (c >> 5)] + bias[c];
    out16[idx] = (half_t)(v > 0.f ? v : 0.f);
}

// layer-2 normalize + relu fused with FC (64 -> 10)
__global__ void norm_relu_fc(const float* __restrict__ agg, const float* __restrict__ den,
                             const float* __restrict__ bias,
                             const float* __restrict__ fc_w, const float* __restrict__ fc_b,
                             float* __restrict__ out, int N)
{
    int nd = blockIdx.x * 256 + threadIdx.x;
    if (nd >= N) return;
    float a[OUT_DIM];
#pragma unroll
    for (int o = 0; o < OUT_DIM; ++o) a[o] = fc_b[o];
    for (int c = 0; c < OUTC; ++c) {
        float v = agg[(size_t)nd * OUTC + c] / den[nd * 2 + (c >> 5)] + bias[c];
        v = v > 0.f ? v : 0.f;
#pragma unroll
        for (int o = 0; o < OUT_DIM; ++o) a[o] += v * fc_w[c * OUT_DIM + o];
    }
#pragma unroll
    for (int o = 0; o < OUT_DIM; ++o) out[(size_t)nd * OUT_DIM + o] = a[o];
}

// ---------------- host launcher ----------------

extern "C" void kernel_launch(void* const* d_in, const int* in_sizes, int n_in,
                              void* d_out, int out_size, void* d_ws, size_t ws_size,
                              hipStream_t stream)
{
    const int*   x        = (const int*)  d_in[0];
    const int*   lengths  = (const int*)  d_in[1];
    const int*   ei       = (const int*)  d_in[2];
    const float* emb      = (const float*)d_in[3];
    const float* w_ih_f   = (const float*)d_in[4];
    const float* w_hh_f   = (const float*)d_in[5];
    const float* b_f      = (const float*)d_in[6];
    const float* w_ih_b   = (const float*)d_in[7];
    const float* b_b      = (const float*)d_in[9];
    const float* gat1_w   = (const float*)d_in[10];
    const float* gat1_as  = (const float*)d_in[11];
    const float* gat1_ad  = (const float*)d_in[12];
    const float* gat1_b   = (const float*)d_in[13];
    const float* gat2_w   = (const float*)d_in[14];
    const float* gat2_as  = (const float*)d_in[15];
    const float* gat2_ad  = (const float*)d_in[16];
    const float* gat2_b   = (const float*)d_in[17];
    const float* fc_w     = (const float*)d_in[18];
    const float* fc_b     = (const float*)d_in[19];
    float* out = (float*)d_out;

    const int N  = in_sizes[1];          // 50000
    const int nE = in_sizes[2] / 2;      // 800000
    const int VOCAB = in_sizes[3] / EMBED;

    // deterministic bump allocator over d_ws
    char* wsp = (char*)d_ws;
    auto alloc = [&](size_t bytes) {
        char* p = wsp; wsp += (bytes + 255) & ~(size_t)255; return p;
    };
    half_t* emb16  = (half_t*)alloc((size_t)VOCAB * EMBED * 2);
    half_t* fih_f  = (half_t*)alloc(32 * 2 * 512 * 2);
    half_t* fhh_f  = (half_t*)alloc(32 * 4 * 512 * 2);
    half_t* fih_b  = (half_t*)alloc(32 * 2 * 512 * 2);
    half_t* fg1    = (half_t*)alloc(4 * 8 * 512 * 2);
    half_t* fg2    = (half_t*)alloc(4 * 2 * 512 * 2);
    half_t* feat16 = (half_t*)alloc((size_t)N * 256 * 2);
    float*  Hbuf   = (float*) alloc((size_t)N * OUTC * 4);
    float*  ssrc   = (float*) alloc((size_t)N * HEADS * 4);
    float*  sdst   = (float*) alloc((size_t)N * HEADS * 4);
    unsigned* mbuf = (unsigned*)alloc((size_t)N * HEADS * 4);
    float*  den    = (float*) alloc((size_t)N * HEADS * 4);
    float*  agg    = (float*) alloc((size_t)N * OUTC * 4);
    half_t* relu16 = (half_t*)alloc((size_t)N * OUTC * 2);

    auto cdiv = [](long a, long b) { return (int)((a + b - 1) / b); };

    // precompute: f16 embedding + fragment-ordered f16 weights
    convert_f16<<<cdiv((long)VOCAB * EMBED, 256), 256, 0, stream>>>(emb, emb16, VOCAB * EMBED);
    pack_b_frags<<<cdiv(32 * 2 * 512, 256), 256, 0, stream>>>(w_ih_f, fih_f, 64,  32, 64,  1);
    pack_b_frags<<<cdiv(32 * 4 * 512, 256), 256, 0, stream>>>(w_hh_f, fhh_f, 128, 32, 128, 1);
    pack_b_frags<<<cdiv(32 * 2 * 512, 256), 256, 0, stream>>>(w_ih_b, fih_b, 64,  32, 64,  1);
    pack_b_frags<<<cdiv(4 * 8 * 512, 256), 256, 0, stream>>>(gat1_w, fg1, 256, 4, 64, 0);
    pack_b_frags<<<cdiv(4 * 2 * 512, 256), 256, 0, stream>>>(gat2_w, fg2, 64,  4, 64, 0);

    // LSTM (fwd masked + bwd single cell) -> feat16 [N][256]
    lstm_kernel<<<N / 16, 256, 0, stream>>>(x, lengths, emb16, fih_f, fhh_f, b_f,
                                            fih_b, b_b, feat16);

    const int totE = nE + N;

    // ---- GAT layer 1 ----
    gat_linear<<<N / 16, 128, 0, stream>>>(feat16, fg1, Hbuf, 256);
    node_scores<<<cdiv((long)N * HEADS, 256), 256, 0, stream>>>(Hbuf, gat1_as, gat1_ad, ssrc, sdst, N);
    fill_u32<<<cdiv((long)N * HEADS, 256), 256, 0, stream>>>(mbuf, (long)N * HEADS);
    fill_u32<<<cdiv((long)N * HEADS, 256), 256, 0, stream>>>((unsigned*)den, (long)N * HEADS);
    fill_u32<<<cdiv((long)N * OUTC, 256), 256, 0, stream>>>((unsigned*)agg, (long)N * OUTC);
    edge_max<<<cdiv(totE, 256), 256, 0, stream>>>(ei, nE, N, ssrc, sdst, mbuf);
    edge_acc<<<cdiv((long)totE * HEADS, 256), 256, 0, stream>>>(ei, nE, N, ssrc, sdst, mbuf, Hbuf, agg, den);
    norm_relu16<<<cdiv((long)N * OUTC, 256), 256, 0, stream>>>(agg, den, gat1_b, relu16, N);

    // ---- GAT layer 2 ----
    gat_linear<<<N / 16, 128, 0, stream>>>(relu16, fg2, Hbuf, 64);
    node_scores<<<cdiv((long)N * HEADS, 256), 256, 0, stream>>>(Hbuf, gat2_as, gat2_ad, ssrc, sdst, N);
    fill_u32<<<cdiv((long)N * HEADS, 256), 256, 0, stream>>>(mbuf, (long)N * HEADS);
    fill_u32<<<cdiv((long)N * HEADS, 256), 256, 0, stream>>>((unsigned*)den, (long)N * HEADS);
    fill_u32<<<cdiv((long)N * OUTC, 256), 256, 0, stream>>>((unsigned*)agg, (long)N * OUTC);
    edge_max<<<cdiv(totE, 256), 256, 0, stream>>>(ei, nE, N, ssrc, sdst, mbuf);
    edge_acc<<<cdiv((long)totE * HEADS, 256), 256, 0, stream>>>(ei, nE, N, ssrc, sdst, mbuf, Hbuf, agg, den);

    // normalize + relu + FC -> d_out [N][10]
    norm_relu_fc<<<cdiv(N, 256), 256, 0, stream>>>(agg, den, gat2_b, fc_w, fc_b, out, N);
}
